// EINet_34883724378340
// MI455X (gfx1250) — compile-verified, matrix operand order checked by reading, not run on hardware
//
#include <hip/hip_runtime.h>
#include <hip/hip_fp16.h>
#include <math.h>

// ---------------------------------------------------------------------------
// EINet (SnowflakeNet-style completion) on gfx1250.
// All linear layers run through a wave32 WMMA GEMM (f16 in, f32 accumulate,
// v_wmma_f32_16x16x32_f16). 4 waves/block, 64x64 block tile, 4 WMMA per wave
// per K-step. A fragments are loaded straight from global (2 contiguous
// 8-float runs per lane -> global_load_b128, no LDS); B is staged in LDS in
// WMMA fragment order so fragment build is two ds_load_b128.
// ---------------------------------------------------------------------------

typedef __attribute__((ext_vector_type(16))) _Float16 v16h;
typedef __attribute__((ext_vector_type(8)))  _Float16 v8h;
typedef __attribute__((ext_vector_type(8)))  float    v8f;

union FragU { v8h p[2]; v16h f; };

// ------------------------------ WMMA GEMM ----------------------------------
// C[rows x cols] = act(A[rows x K] @ W[K x cols] + bias[col % bias_mod])
// blockDim = 128 (4 waves). Wave w owns rows [blockIdx.x*64 + w*16, +16).
// Block owns cols [blockIdx.y*64, +64) as four 16-wide WMMA tiles.
// ISA 7.12.2 16-bit fragment layout: element (m|n, k) lives at
//   lane = (m&15) | (((k>>3)&1)<<4),  elem j = ((k>>4)<<3) | (k&7)
__global__ void k_gemm(const float* __restrict__ A, const float* __restrict__ W,
                       const float* __restrict__ bias, float* __restrict__ C,
                       int rows, int K, int cols, int bias_mod, int act) {
  __shared__ _Float16 BF[4][32][16];   // [colTile][lane][elem]
  const int tid  = threadIdx.x;
  const int w    = tid >> 5;           // wave id 0..3
  const int lane = tid & 31;
  const int half = lane >> 4;
  const int mr   = lane & 15;
  const int rowBase = blockIdx.x * 64;
  const int col0    = blockIdx.y * 64;
  const int waveRow = rowBase + w * 16;
  const int gm      = waveRow + mr;    // the A row this lane owns
  const float* arow = A + (long)gm * K;

  const bool fastA = (rowBase + 64 <= rows) && ((K & 3) == 0);
  const bool fastB = (col0 + 64 <= cols) && ((cols & 3) == 0);

  v8f acc[4] = {};

  for (int kk = 0; kk < K; kk += 32) {
    const bool fullK = (kk + 32 <= K);
    __syncthreads();
    // ---- stage B (32 k x 64 cols) into LDS, fragment-ordered ----
    if (fastB && fullK) {
      // 512 float4 quads / 128 threads = 4 per thread; coalesced b128 reads.
#pragma unroll
      for (int i = 0; i < 4; ++i) {
        int e  = tid + 128 * i;
        int k  = e >> 4;                 // 0..31
        int cq = (e & 15) << 2;          // 0,4,...,60 (stays in one col tile)
        const float4 f = *(const float4*)&W[(long)(kk + k) * cols + col0 + cq];
        int j   = ((k >> 4) << 3) | (k & 7);
        int hb  = ((k >> 3) & 1) << 4;
        int ct  = cq >> 4, cl = cq & 15;
        BF[ct][(cl + 0) | hb][j] = (_Float16)f.x;
        BF[ct][(cl + 1) | hb][j] = (_Float16)f.y;
        BF[ct][(cl + 2) | hb][j] = (_Float16)f.z;
        BF[ct][(cl + 3) | hb][j] = (_Float16)f.w;
      }
    } else {
      // checked scalar path: 2048 elements / 128 threads = 16 per thread
#pragma unroll
      for (int i = 0; i < 16; ++i) {
        int e = tid + 128 * i;
        int k = e >> 6, col = e & 63;
        int gk = kk + k, gc = col0 + col;
        bool ok = (gk < K) && (gc < cols);
        long ii = ok ? ((long)gk * cols + gc) : 0;
        float v = W[ii];
        int lane_ = (col & 15) | (((k >> 3) & 1) << 4);
        int j     = ((k >> 4) << 3) | (k & 7);
        BF[col >> 4][lane_][j] = ok ? (_Float16)v : (_Float16)0.f;
      }
    }
    if (kk + 32 < K) {  // hint next K tile -> global_prefetch_b8
      __builtin_prefetch(&arow[kk + 32], 0, 1);
      __builtin_prefetch(&W[(long)(kk + 32) * cols + col0], 0, 1);
    }
    __syncthreads();
    // ---- A fragment straight from global: two 8-float contiguous runs ----
    v16h a;
    if (fastA && fullK) {
      const float4* p0 = (const float4*)&arow[kk + (half << 3)];
      const float4* p1 = (const float4*)&arow[kk + (half << 3) + 16];
      float4 f0 = p0[0], f1 = p0[1], f2 = p1[0], f3 = p1[1];
      a[0]  = (_Float16)f0.x; a[1]  = (_Float16)f0.y;
      a[2]  = (_Float16)f0.z; a[3]  = (_Float16)f0.w;
      a[4]  = (_Float16)f1.x; a[5]  = (_Float16)f1.y;
      a[6]  = (_Float16)f1.z; a[7]  = (_Float16)f1.w;
      a[8]  = (_Float16)f2.x; a[9]  = (_Float16)f2.y;
      a[10] = (_Float16)f2.z; a[11] = (_Float16)f2.w;
      a[12] = (_Float16)f3.x; a[13] = (_Float16)f3.y;
      a[14] = (_Float16)f3.z; a[15] = (_Float16)f3.w;
    } else {
#pragma unroll
      for (int j = 0; j < 16; ++j) {
        int k = kk + ((j >> 3) << 4) + (half << 3) + (j & 7);
        bool ok = (gm < rows) && (k < K);
        long ii = ok ? ((long)gm * K + k) : 0;
        float v = A[ii];
        a[j] = ok ? (_Float16)v : (_Float16)0.f;
      }
    }
    // ---- 4 WMMA per wave; B fragments are two b128 LDS reads each ----
#pragma unroll
    for (int t = 0; t < 4; ++t) {
      FragU ub;
      const v8h* bp = (const v8h*)&BF[t][lane][0];
      ub.p[0] = bp[0]; ub.p[1] = bp[1];
      acc[t] = __builtin_amdgcn_wmma_f32_16x16x32_f16(
          /*neg_a=*/false, a, /*neg_b=*/false, ub.f,
          /*c_mod=*/(short)0, acc[t], /*reuse_a=*/false, /*reuse_b=*/false);
    }
  }
  // ---- epilogue: D layout VGPR r -> M = r + 8*half, N = lane&15 ----
#pragma unroll
  for (int t = 0; t < 4; ++t) {
    int gc = col0 + t * 16 + mr;
    if (gc >= cols) continue;
    float bval = bias ? bias[gc % bias_mod] : 0.f;
#pragma unroll
    for (int r = 0; r < 8; ++r) {
      int gmr = waveRow + r + (half << 3);
      if (gmr < rows) {
        float v = acc[t][r] + bval;
        if (act == 1) v = fmaxf(v, 0.f);
        C[(long)gmr * cols + gc] = v;
      }
    }
  }
}

// ----------------------- farthest point sampling ---------------------------
__global__ void k_fps(const float* __restrict__ xyz, int n, int npoint,
                      int* __restrict__ out) {
  __shared__ float sd[2048];
  __shared__ float rv[256];
  __shared__ int   ri[256];
  __shared__ int   s_last;
  const int b = blockIdx.x;
  const float* p = xyz + (long)b * n * 3;
  int* o = out + (long)b * npoint;
  for (int i = threadIdx.x; i < n; i += blockDim.x) sd[i] = 1e10f;
  if (threadIdx.x == 0) { o[0] = 0; s_last = 0; }
  __syncthreads();
  for (int t = 1; t < npoint; ++t) {
    const int last = s_last;
    const float lx = p[last * 3], ly = p[last * 3 + 1], lz = p[last * 3 + 2];
    float bestv = -1.f; int besti = 0;
    for (int i = threadIdx.x; i < n; i += blockDim.x) {
      float dx = p[i * 3] - lx, dy = p[i * 3 + 1] - ly, dz = p[i * 3 + 2] - lz;
      float m = fminf(sd[i], dx * dx + dy * dy + dz * dz);
      sd[i] = m;
      if (m > bestv) { bestv = m; besti = i; }
    }
    rv[threadIdx.x] = bestv; ri[threadIdx.x] = besti;
    __syncthreads();
    for (int s = blockDim.x / 2; s > 0; s >>= 1) {
      if (threadIdx.x < s) {
        if (rv[threadIdx.x + s] > rv[threadIdx.x] ||
            (rv[threadIdx.x + s] == rv[threadIdx.x] &&
             ri[threadIdx.x + s] < ri[threadIdx.x])) {
          rv[threadIdx.x] = rv[threadIdx.x + s];
          ri[threadIdx.x] = ri[threadIdx.x + s];
        }
      }
      __syncthreads();
    }
    if (threadIdx.x == 0) { o[t] = ri[0]; s_last = ri[0]; }
    __syncthreads();
  }
}

// ------------------------------ brute KNN ----------------------------------
__global__ void k_knn(const float* __restrict__ ref, const float* __restrict__ query,
                      int B, int n, int m, int K,
                      int* __restrict__ idx, float* __restrict__ d2o) {
  long tid = (long)blockIdx.x * blockDim.x + threadIdx.x;
  if (tid >= (long)B * m) return;
  const int b = (int)(tid / m);
  const float* r = ref + (long)b * n * 3;
  const float qx = query[tid * 3], qy = query[tid * 3 + 1], qz = query[tid * 3 + 2];
  float bd[24]; int bi[24];
  for (int t = 0; t < K; ++t) { bd[t] = 1e30f; bi[t] = 0; }
  for (int i = 0; i < n; ++i) {
    float dx = r[i * 3] - qx, dy = r[i * 3 + 1] - qy, dz = r[i * 3 + 2] - qz;
    float d = dx * dx + dy * dy + dz * dz;
    if (d < bd[K - 1]) {
      int t = K - 1;
      while (t > 0 && bd[t - 1] > d) { bd[t] = bd[t - 1]; bi[t] = bi[t - 1]; --t; }
      bd[t] = d; bi[t] = i;
    }
  }
  for (int t = 0; t < K; ++t) { idx[tid * K + t] = bi[t]; d2o[tid * K + t] = bd[t]; }
}

// ------------------------------ small ops ----------------------------------
__global__ void k_gather_rows(const float* __restrict__ src, const int* __restrict__ idx,
                              float* __restrict__ out, int B, int n, int m, int C) {
  long tid = (long)blockIdx.x * blockDim.x + threadIdx.x;
  if (tid >= (long)B * m * C) return;
  int c = (int)(tid % C); long r = tid / C;
  int j = (int)(r % m);  int b = (int)(r / m);
  out[tid] = src[((long)b * n + idx[(long)b * m + j]) * C + c];
}

__global__ void k_group_sa(const float* __restrict__ xyz, const float* __restrict__ fea,
                           const float* __restrict__ nxyz, const int* __restrict__ idx,
                           float* __restrict__ out, int B, int n, int m, int K, int cf) {
  long tid = (long)blockIdx.x * blockDim.x + threadIdx.x;
  if (tid >= (long)B * m * K) return;
  long rem = tid % ((long)m * K);
  int b = (int)(tid / ((long)m * K));
  int j = (int)(rem / K);
  const int id = idx[tid];
  const float* xs = xyz + ((long)b * n + id) * 3;
  const float* nx = nxyz + ((long)b * m + j) * 3;
  float* o = out + tid * (3 + cf);
  o[0] = xs[0] - nx[0]; o[1] = xs[1] - nx[1]; o[2] = xs[2] - nx[2];
  const float* fs = fea + ((long)b * n + id) * cf;
  for (int c = 0; c < cf; ++c) o[3 + c] = fs[c];
}

__global__ void k_maxpool(const float* __restrict__ in, float* __restrict__ out,
                          long R, int K, int C) {
  long tid = (long)blockIdx.x * blockDim.x + threadIdx.x;
  if (tid >= R * C) return;
  int c = (int)(tid % C); long r = tid / C;
  float m = -1e30f;
  for (int t = 0; t < K; ++t) m = fmaxf(m, in[(r * K + t) * C + c]);
  out[tid] = m;
}

__global__ void k_max_over_mid(const float* __restrict__ in, float* __restrict__ out,
                               int B, int n, int C) {
  long tid = (long)blockIdx.x * blockDim.x + threadIdx.x;
  if (tid >= (long)B * C) return;
  int c = (int)(tid % C); int b = (int)(tid / C);
  float m = -1e30f;
  for (int i = 0; i < n; ++i) m = fmaxf(m, in[((long)b * n + i) * C + c]);
  out[tid] = m;
}

// out[b,j,t,c] = A[b,j,c] - Bsrc[b, idx[b,j,t], c]
__global__ void k_gather_sub(const float* __restrict__ A, const float* __restrict__ Bsrc,
                             const int* __restrict__ idx, float* __restrict__ out,
                             int B, int n, int m, int K, int C) {
  long tid = (long)blockIdx.x * blockDim.x + threadIdx.x;
  if (tid >= (long)B * m * K * C) return;
  int c = (int)(tid % C); long r = tid / C;
  long bj = r / K;
  int j = (int)(bj % m); int b = (int)(bj / m);
  out[tid] = A[((long)b * m + j) * C + c] - Bsrc[((long)b * n + idx[r]) * C + c];
}

__global__ void k_add(const float* __restrict__ a, const float* __restrict__ b,
                      float* __restrict__ o, long n) {
  long tid = (long)blockIdx.x * blockDim.x + threadIdx.x;
  if (tid < n) o[tid] = a[tid] + b[tid];
}

// out[r,t,c] = per_row[r,c] + per_nbr[r,t,c]
__global__ void k_bcast_add(const float* __restrict__ pr, const float* __restrict__ pn,
                            float* __restrict__ out, long R, int K, int C) {
  long tid = (long)blockIdx.x * blockDim.x + threadIdx.x;
  if (tid >= R * (long)K * C) return;
  int c = (int)(tid % C); long rk = tid / C; long r = rk / K;
  out[tid] = pr[r * C + c] + pn[tid];
}

__global__ void k_softmax_axis(float* __restrict__ x, long R, int K, int C) {
  long tid = (long)blockIdx.x * blockDim.x + threadIdx.x;
  if (tid >= R * C) return;
  int c = (int)(tid % C); long r = tid / C;
  float mx = -1e30f;
  for (int t = 0; t < K; ++t) mx = fmaxf(mx, x[(r * K + t) * C + c]);
  float s = 0.f;
  for (int t = 0; t < K; ++t) {
    float e = expf(x[(r * K + t) * C + c] - mx);
    x[(r * K + t) * C + c] = e; s += e;
  }
  float inv = 1.f / s;
  for (int t = 0; t < K; ++t) x[(r * K + t) * C + c] *= inv;
}

__global__ void k_attn_agg(const float* __restrict__ attn, const float* __restrict__ val,
                           float* __restrict__ out, long R, int K, int C) {
  long tid = (long)blockIdx.x * blockDim.x + threadIdx.x;
  if (tid >= R * C) return;
  int c = (int)(tid % C); long r = tid / C;
  float s = 0.f;
  for (int t = 0; t < K; ++t) s += attn[(r * K + t) * C + c] * val[(r * K + t) * C + c];
  out[tid] = s;
}

__global__ void k_concat2(const float* __restrict__ a, int ca,
                          const float* __restrict__ b, int cb,
                          float* __restrict__ out, long rows) {
  int ct = ca + cb;
  long tid = (long)blockIdx.x * blockDim.x + threadIdx.x;
  if (tid >= rows * ct) return;
  int c = (int)(tid % ct); long r = tid / ct;
  out[tid] = (c < ca) ? a[r * ca + c] : b[r * cb + (c - ca)];
}

__global__ void k_concat3(const float* __restrict__ a, int ca,
                          const float* __restrict__ b, int cb,
                          const float* __restrict__ d, int cd,
                          float* __restrict__ out, long rows) {
  int ct = ca + cb + cd;
  long tid = (long)blockIdx.x * blockDim.x + threadIdx.x;
  if (tid >= rows * ct) return;
  int c = (int)(tid % ct); long r = tid / ct;
  out[tid] = (c < ca) ? a[r * ca + c]
           : (c < ca + cb) ? b[r * cb + (c - ca)]
                           : d[r * cd + (c - ca - cb)];
}

// concat of per-row features with a per-batch broadcast vector
__global__ void k_concat_bcast(const float* __restrict__ a, int ca,
                               const float* __restrict__ g, int cb,
                               float* __restrict__ out, long rows, int rowsPerB) {
  int ct = ca + cb;
  long tid = (long)blockIdx.x * blockDim.x + threadIdx.x;
  if (tid >= rows * ct) return;
  int c = (int)(tid % ct); long r = tid / ct;
  out[tid] = (c < ca) ? a[r * ca + c] : g[(r / rowsPerB) * cb + (c - ca)];
}

__global__ void k_copy(float* __restrict__ dst, const float* __restrict__ src, long n) {
  long tid = (long)blockIdx.x * blockDim.x + threadIdx.x;
  if (tid < n) dst[tid] = src[tid];
}

__global__ void k_copy_block(float* __restrict__ dst, long dstStride, long dstOff,
                             const float* __restrict__ src, long srcStride,
                             int B, long epb) {
  long tid = (long)blockIdx.x * blockDim.x + threadIdx.x;
  if (tid >= (long)B * epb) return;
  long b = tid / epb, e = tid % epb;
  dst[b * dstStride + dstOff + e] = src[b * srcStride + e];
}

__global__ void k_interp(const float* __restrict__ sf, const int* __restrict__ idx,
                         const float* __restrict__ d2, float* __restrict__ out,
                         int B, int Sn, int m, int K, int C) {
  long tid = (long)blockIdx.x * blockDim.x + threadIdx.x;
  if (tid >= (long)B * m * C) return;
  int c = (int)(tid % C); long r = tid / C;
  int b = (int)(r / m);
  const int* id = idx + r * K;
  const float* dd = d2 + r * K;
  float ws = 0.f, acc = 0.f;
  for (int t = 0; t < K; ++t) {
    float w = 1.f / (sqrtf(fmaxf(dd[t], 0.f)) + 1e-8f);
    ws += w;
    acc += w * sf[((long)b * Sn + id[t]) * C + c];
  }
  out[tid] = acc / ws;
}

__global__ void k_tanh_scale(float* __restrict__ x, float s, long n) {
  long tid = (long)blockIdx.x * blockDim.x + threadIdx.x;
  if (tid < n) x[tid] = tanhf(x[tid]) * s;
}

__global__ void k_repeat_add(const float* __restrict__ pcd, const float* __restrict__ off,
                             float* __restrict__ out, int B, int n, int up) {
  long tid = (long)blockIdx.x * blockDim.x + threadIdx.x;
  long total = (long)B * n * up * 3;
  if (tid >= total) return;
  int d = (int)(tid % 3); long r = tid / 3;
  long nu = (long)n * up;
  int iu = (int)(r % nu); int b = (int)(r / nu);
  int i = iu / up;
  out[tid] = pcd[((long)b * n + i) * 3 + d] + off[tid];
}

// cons2[b, d, colOff + i] = off[b, i, d]
__global__ void k_off_to_cons(const float* __restrict__ off, float* __restrict__ cons,
                              int B, int n, int totalCols, int colOff) {
  long tid = (long)blockIdx.x * blockDim.x + threadIdx.x;
  if (tid >= (long)B * n * 3) return;
  int d = (int)(tid % 3); long r = tid / 3;
  int i = (int)(r % n); int b = (int)(r / n);
  cons[((long)b * 3 + d) * totalCols + colOff + i] = off[tid];
}

// ------------------------------ host side ----------------------------------
static inline unsigned gsz(long total, int bs) {
  return (unsigned)((total + bs - 1) / bs);
}

extern "C" void kernel_launch(void* const* d_in, const int* in_sizes, int n_in,
                              void* d_out, int out_size, void* d_ws, size_t ws_size,
                              hipStream_t stream) {
  (void)out_size; (void)ws_size;
  // ---- input leaf mapping: jax tree-flatten order (dict keys sorted) ------
  // params: ex_cons(0-3) ex_mlp(4-7) ex_xyz(8-11) ps_b(12) ps_w(13)
  //         sa1(14-17) sa2(18-21) sa4(22-25) t1(26-43) t2(44-61) up(62-91)
  // partial_cloud: found by size scan (8*2048*3 is unique)
  auto F = [&](int i) { return (const float*)d_in[i]; };
  int pcIdx = n_in - 1;
  for (int i = 0; i < n_in; ++i)
    if (in_sizes[i] == 8 * 2048 * 3) { pcIdx = i; break; }
  const float* xyz = F(pcIdx);
  float* out = (float*)d_out;

  // output regions
  const long o_p0 = 0, o_p1 = 24576, o_p2 = 49152, o_p3 = 98304;
  const long o_c1 = 491520, o_c2 = 495616;
  const int consCols = 19456;

  // ---- workspace bump allocator ----
  size_t cur = 0;
  auto alloc  = [&](size_t nf) { size_t o = cur; cur += ((nf * 4 + 255) / 256) * 256;
                                 return (float*)((char*)d_ws + o); };
  auto alloci = [&](size_t ni) { size_t o = cur; cur += ((ni * 4 + 255) / 256) * 256;
                                 return (int*)((char*)d_ws + o); };
  auto gemm = [&](const float* A, const float* W, const float* bias, float* C,
                  long rows, int K, int cols, int act, int bias_mod = 0) {
    if (bias_mod == 0) bias_mod = cols;
    dim3 g((unsigned)((rows + 63) / 64), (unsigned)((cols + 63) / 64), 1);
    k_gemm<<<g, 128, 0, stream>>>(A, W, bias, C, (int)rows, K, cols, bias_mod, act);
  };

  // persistent buffers
  float* l1_xyz   = alloc((size_t)8 * 512 * 3);
  float* l1_fea   = alloc((size_t)8 * 512 * 128);
  float* l2_xyz   = alloc((size_t)8 * 128 * 3);
  float* l2_fea   = alloc((size_t)8 * 128 * 256);
  float* glo      = alloc((size_t)8 * 512);
  float* hbuf     = alloc((size_t)8 * 512 * 128);
  float* seed     = alloc((size_t)8 * 1024 * 3);
  float* seed_fea = alloc((size_t)8 * 1024 * 128);
  float* childBuf = alloc((size_t)8 * 16384 * 128);
  const size_t mark = cur;

  // ---- SA block (FPS + KNN group + shared MLP + maxpool) ----
  auto sa = [&](const float* x_in, int n, const float* f_in, int cf, int npoint,
                const float* w0, const float* b0, int h0,
                const float* w1, const float* b1, int c1,
                float* oxyz, float* ofea) {
    cur = mark;
    int* fpsI = alloci((size_t)8 * npoint);
    k_fps<<<8, 256, 0, stream>>>(x_in, n, npoint, fpsI);
    k_gather_rows<<<gsz((long)8 * npoint * 3, 256), 256, 0, stream>>>(
        x_in, fpsI, oxyz, 8, n, npoint, 3);
    int* knnI = alloci((size_t)8 * npoint * 16);
    float* knnD = alloc((size_t)8 * npoint * 16);
    k_knn<<<gsz((long)8 * npoint, 128), 128, 0, stream>>>(
        x_in, oxyz, 8, n, npoint, 16, knnI, knnD);
    long Rg = (long)8 * npoint * 16;
    float* grp = alloc(Rg * (3 + cf));
    k_group_sa<<<gsz(Rg, 256), 256, 0, stream>>>(
        x_in, f_in, oxyz, knnI, grp, 8, n, npoint, 16, cf);
    float* h1 = alloc(Rg * h0);
    gemm(grp, w0, b0, h1, Rg, 3 + cf, h0, 1);
    float* h2 = alloc(Rg * c1);
    gemm(h1, w1, b1, h2, Rg, h0, c1, 0);
    k_maxpool<<<gsz((long)8 * npoint * c1, 256), 256, 0, stream>>>(
        h2, ofea, (long)8 * npoint, 16, c1);
  };

  // ---- point transformer (vector attention over KNN) ----
  auto xform = [&](const float* x_pts, float* fea, int n, int C, int pb) {
    cur = mark;
    const float* attn0W = F(pb + 0);  const float* attn0B = F(pb + 1);
    const float* attn1W = F(pb + 2);  const float* attn1B = F(pb + 3);
    const float* endW   = F(pb + 4);  const float* endB   = F(pb + 5);
    const float* kW     = F(pb + 6);  const float* kB     = F(pb + 7);
    const float* pos0W  = F(pb + 8);  const float* pos0B  = F(pb + 9);
    const float* pos1W  = F(pb + 10); const float* pos1B  = F(pb + 11);
    const float* qW     = F(pb + 12); const float* qB     = F(pb + 13);
    const float* startW = F(pb + 14); const float* startB = F(pb + 15);
    const float* vW     = F(pb + 16); const float* vB     = F(pb + 17);
    long R = (long)8 * n, RK = R * 16;
    float* xs = alloc(R * 64);
    gemm(fea, startW, startB, xs, R, C, 64, 0);
    int* tI = alloci(RK); float* tD = alloc(RK);
    k_knn<<<gsz(R, 128), 128, 0, stream>>>(x_pts, x_pts, 8, n, n, 16, tI, tD);
    float* q = alloc(R * 64); float* kf = alloc(R * 64); float* v = alloc(R * 64);
    gemm(xs, qW, qB, q, R, 64, 64, 0);
    gemm(xs, kW, kB, kf, R, 64, 64, 0);
    gemm(xs, vW, vB, v, R, 64, 64, 0);
    float* qkrel = alloc(RK * 64);
    k_gather_sub<<<gsz(RK * 64, 256), 256, 0, stream>>>(q, kf, tI, qkrel, 8, n, n, 16, 64);
    float* posrel = alloc(RK * 3);
    k_gather_sub<<<gsz(RK * 3, 256), 256, 0, stream>>>(x_pts, x_pts, tI, posrel, 8, n, n, 16, 3);
    float* posh = alloc(RK * 64);
    gemm(posrel, pos0W, pos0B, posh, RK, 3, 64, 1);
    float* pose = alloc(RK * 64);
    gemm(posh, pos1W, pos1B, pose, RK, 64, 64, 0);
    k_add<<<gsz(RK * 64, 256), 256, 0, stream>>>(qkrel, pose, qkrel, RK * 64);
    float* ah = alloc(RK * 256);
    gemm(qkrel, attn0W, attn0B, ah, RK, 64, 256, 1);
    float* attn = alloc(RK * 64);
    gemm(ah, attn1W, attn1B, attn, RK, 256, 64, 0);
    k_softmax_axis<<<gsz(R * 64, 256), 256, 0, stream>>>(attn, R, 16, 64);
    float* val = alloc(RK * 64);
    k_bcast_add<<<gsz(RK * 64, 256), 256, 0, stream>>>(v, pose, val, R, 16, 64);
    float* agg = alloc(R * 64);
    k_attn_agg<<<gsz(R * 64, 256), 256, 0, stream>>>(attn, val, agg, R, 16, 64);
    float* eo = alloc(R * C);
    gemm(agg, endW, endB, eo, R, 64, C, 0);
    k_add<<<gsz(R * C, 256), 256, 0, stream>>>(fea, eo, fea, R * C);
  };

  // ======================= FeaExtract =======================
  sa(xyz, 2048, xyz, 3, 512, F(14), F(15), 32, F(16), F(17), 128, l1_xyz, l1_fea);
  xform(l1_xyz, l1_fea, 512, 128, 26);
  sa(l1_xyz, 512, l1_fea, 128, 128, F(18), F(19), 128, F(20), F(21), 256, l2_xyz, l2_fea);
  xform(l2_xyz, l2_fea, 128, 256, 44);

  // ======================= global feature + seeds =======================
  cur = mark;
  {
    long R2 = 8 * 128;
    float* s4in = alloc(R2 * 259);
    k_concat2<<<gsz(R2 * 259, 256), 256, 0, stream>>>(l2_xyz, 3, l2_fea, 256, s4in, R2);
    float* s4h = alloc(R2 * 512);
    gemm(s4in, F(22), F(23), s4h, R2, 259, 512, 1);
    float* s4o = alloc(R2 * 512);
    gemm(s4h, F(24), F(25), s4o, R2, 512, 512, 0);
    k_max_over_mid<<<gsz((long)8 * 512, 256), 256, 0, stream>>>(s4o, glo, 8, 128, 512);
    // x1 = glo @ ps_w(512 x 65536) + ps_b  (bias repeats per 128-wide chunk)
    float* x1 = alloc((size_t)8 * 65536);
    gemm(glo, F(13), F(12), x1, 8, 512, 65536, 0, /*bias_mod=*/128);
    long RS = (long)8 * 512;
    float* exin = alloc(RS * 640);
    k_concat_bcast<<<gsz(RS * 640, 256), 256, 0, stream>>>(x1, 128, glo, 512, exin, RS, 512);
    float* exh = alloc(RS * 256);
    gemm(exin, F(4), F(5), exh, RS, 640, 256, 1);
    gemm(exh, F(6), F(7), hbuf, RS, 256, 128, 0);       // h
    float* exx = alloc(RS * 64);
    gemm(hbuf, F(8), F(9), exx, RS, 128, 64, 1);
    float* u0 = alloc(RS * 3);
    gemm(exx, F(10), F(11), u0, RS, 64, 3, 0);          // u0_xyz
    float* exc = alloc(RS * 64);
    gemm(hbuf, F(0), F(1), exc, RS, 128, 64, 1);
    gemm(exc, F(2), F(3), out + o_c1, RS, 64, 1, 0);    // cons1 -> d_out
    // seed / seed_fea (point-axis concat)
    k_copy_block<<<gsz((long)8 * 512 * 3, 256), 256, 0, stream>>>(
        seed, 1024 * 3, 0, l1_xyz, 512 * 3, 8, 512 * 3);
    k_copy_block<<<gsz((long)8 * 512 * 3, 256), 256, 0, stream>>>(
        seed, 1024 * 3, 512 * 3, u0, 512 * 3, 8, 512 * 3);
    k_copy_block<<<gsz((long)8 * 512 * 128, 256), 256, 0, stream>>>(
        seed_fea, 1024 * 128, 0, l1_fea, 512 * 128, 8, 512 * 128);
    k_copy_block<<<gsz((long)8 * 512 * 128, 256), 256, 0, stream>>>(
        seed_fea, 1024 * 128, 512 * 128, hbuf, 512 * 128, 8, 512 * 128);
    k_copy<<<gsz(24576, 256), 256, 0, stream>>>(out + o_p0, seed, 24576); // preds[0]
  }

  // ======================= UPProcess decoder =======================
  const int   ups[3]    = {1, 2, 8};
  const float scales[3] = {0.4f, 0.3f, 0.2f};
  const int   nIn[3]    = {1024, 1024, 2048};
  const long  pOff[3]   = {o_p1, o_p2, o_p3};
  const int   cOff[3]   = {0, 1024, 3072};
  const float* Kprev = seed_fea;
  const float* pcd = seed;
  for (int L = 0; L < 3; ++L) {
    cur = mark;
    const int pb = 62 + 10 * L;
    const int n = nIn[L], up = ups[L], nOut = n * up;
    long R = (long)8 * n, Ro = (long)8 * nOut;
    int* kI = alloci(R * 20); float* kD = alloc(R * 20);
    k_knn<<<gsz(R, 128), 128, 0, stream>>>(seed, pcd, 8, 1024, n, 20, kI, kD);
    float* interp = alloc(R * 128);
    k_interp<<<gsz(R * 128, 256), 256, 0, stream>>>(
        seed_fea, kI, kD, interp, 8, 1024, n, 20, 128);
    float* m1in = alloc(R * 259);
    k_concat3<<<gsz(R * 259, 256), 256, 0, stream>>>(
        pcd, 3, interp, 128, Kprev, 128, m1in, R);
    float* m1h = alloc(R * 256);
    gemm(m1in, F(pb + 4), F(pb + 5), m1h, R, 259, 256, 1);   // mlp1 L0 (relu)
    float* feat = alloc(R * 128);
    gemm(m1h, F(pb + 6), F(pb + 7), feat, R, 256, 128, 1);   // mlp1 L1 + outer relu
    // ps split: (R x up*128) row-major == (B, n*up, 128)
    gemm(feat, F(pb + 8), F(pb + 9), childBuf, R, 128, up * 128, 1);
    float* dh = alloc(Ro * 64);
    gemm(childBuf, F(pb + 0), F(pb + 1), dh, Ro, 128, 64, 1);
    float* off = alloc(Ro * 3);
    gemm(dh, F(pb + 2), F(pb + 3), off, Ro, 64, 3, 0);
    k_tanh_scale<<<gsz(Ro * 3, 256), 256, 0, stream>>>(off, scales[L], Ro * 3);
    float* pcd_new = out + pOff[L];
    k_repeat_add<<<gsz(Ro * 3, 256), 256, 0, stream>>>(pcd, off, pcd_new, 8, n, up);
    k_off_to_cons<<<gsz(Ro * 3, 256), 256, 0, stream>>>(
        off, out + o_c2, 8, nOut, consCols, cOff[L]);
    pcd = pcd_new;
    Kprev = childBuf;
  }
}
